// PegSolitaireGNN_36524401885974
// MI455X (gfx1250) — matrix operand order repaired and checked by reading, not compiled, not run on hardware
//
#include <hip/hip_runtime.h>
#include <hip/hip_bf16.h>
#include <math.h>

#define DEV_INLINE __device__ __forceinline__

// ---------------- problem constants ----------------
constexpr int B_    = 4096;
constexpr int NPG_  = 33;
constexpr int N_    = B_ * NPG_;       // 135168 nodes (divisible by 16)
constexpr int E_    = 409600;
constexpr int ETOT_ = E_ + N_;         // edges + self loops = 544768
constexpr int HC_   = 256;             // layer-1 output width (4 heads x 64)
constexpr int C_    = 64;              // layer-2 width
constexpr int NA_   = 196;             // actions
constexpr float NEG_SLOPE_ = 0.2f;

// ---------------- WMMA vector types ----------------
typedef __attribute__((ext_vector_type(16))) __bf16 v16bf;
typedef __attribute__((ext_vector_type(8)))  float  v8f;

union FragBF { v16bf v; uint4 u[2]; };

// monotonic float<->uint encoding so atomicMax(uint) == float max
DEV_INLINE unsigned fenc(float f) {
    unsigned u = __float_as_uint(f);
    return (u & 0x80000000u) ? ~u : (u | 0x80000000u);
}
DEV_INLINE float fdec(unsigned u) {
    return __uint_as_float((u & 0x80000000u) ? (u & 0x7FFFFFFFu) : ~u);
}
// fenc(-inf) = 0x007FFFFF : init value for the atomic-max buffers
constexpr unsigned NEG_INF_ENC = 0x007FFFFFu;

DEV_INLINE float lrelu(float v) { return v > 0.f ? v : NEG_SLOPE_ * v; }
DEV_INLINE float eluf(float v)  { return v > 0.f ? v : expm1f(v); }

// ---------------- utility kernels ----------------
__global__ void k_zero_f32(float* __restrict__ p, int n) {
    int i = blockIdx.x * blockDim.x + threadIdx.x;
    if (i < n) p[i] = 0.f;
}
__global__ void k_init_u32(unsigned* __restrict__ p, unsigned v, int n) {
    int i = blockIdx.x * blockDim.x + threadIdx.x;
    if (i < n) p[i] = v;
}

// ---------------- layer 1: input projection (K=4, scalar) ----------------
__global__ void k_lin1(const float* __restrict__ x,
                       const float* __restrict__ Wl, const float* __restrict__ Wr,
                       float* __restrict__ xl, float* __restrict__ xr) {
    int node = blockIdx.x;          // N_ blocks
    int col  = threadIdx.x;         // 256 threads
    float x0 = x[node * 4 + 0], x1 = x[node * 4 + 1];
    float x2 = x[node * 4 + 2], x3 = x[node * 4 + 3];
    float l = x0 * Wl[col] + x1 * Wl[256 + col] + x2 * Wl[512 + col] + x3 * Wl[768 + col];
    float r = x0 * Wr[col] + x1 * Wr[256 + col] + x2 * Wr[512 + col] + x3 * Wr[768 + col];
    xl[(size_t)node * HC_ + col] = l;
    xr[(size_t)node * HC_ + col] = r;
}

// ---------------- layer 1 edge softmax: pass A (scores + seg-max) ----------------
__global__ void k_edge1_a(const int* __restrict__ ei,
                          const float* __restrict__ xl, const float* __restrict__ xr,
                          const float* __restrict__ att,
                          float* __restrict__ e1, unsigned* __restrict__ emax) {
    __shared__ float satt[HC_];
    satt[threadIdx.x] = att[threadIdx.x];
    __syncthreads();
    int idx = blockIdx.x * blockDim.x + threadIdx.x;
    if (idx >= ETOT_) return;
    int s, d;
    if (idx < E_) { s = ei[idx]; d = ei[E_ + idx]; } else { s = d = idx - E_; }
    const float* pl = xl + (size_t)s * HC_;
    const float* pr = xr + (size_t)d * HC_;
    for (int h = 0; h < 4; ++h) {
        float acc = 0.f;
        for (int c = 0; c < 64; ++c) {
            float m = lrelu(pl[h * 64 + c] + pr[h * 64 + c]);
            acc += satt[h * 64 + c] * m;
        }
        e1[(size_t)idx * 4 + h] = acc;
        atomicMax(emax + (size_t)d * 4 + h, fenc(acc));
    }
}

// ---------------- layer 1 pass B: exp + seg-sum ----------------
__global__ void k_edge1_b(const int* __restrict__ ei,
                          float* __restrict__ e1, const unsigned* __restrict__ emax,
                          float* __restrict__ denom) {
    int idx = blockIdx.x * blockDim.x + threadIdx.x;
    if (idx >= ETOT_) return;
    int d = (idx < E_) ? ei[E_ + idx] : (idx - E_);
    for (int h = 0; h < 4; ++h) {
        float e = e1[(size_t)idx * 4 + h];
        float ex = expf(e - fdec(emax[(size_t)d * 4 + h]));
        e1[(size_t)idx * 4 + h] = ex;
        atomicAdd(denom + (size_t)d * 4 + h, ex);
    }
}

// ---------------- layer 1 pass C: weighted scatter ----------------
__global__ void k_edge1_c(const int* __restrict__ ei,
                          const float* __restrict__ e1, const float* __restrict__ denom,
                          const float* __restrict__ xl, float* __restrict__ out) {
    int idx = blockIdx.x * blockDim.x + threadIdx.x;
    if (idx >= ETOT_) return;
    int s, d;
    if (idx < E_) { s = ei[idx]; d = ei[E_ + idx]; } else { s = d = idx - E_; }
    const float* pl = xl + (size_t)s * HC_;
    float* po = out + (size_t)d * HC_;
    for (int h = 0; h < 4; ++h) {
        float alpha = e1[(size_t)idx * 4 + h] / denom[(size_t)d * 4 + h];
        for (int c = 0; c < 64; ++c)
            atomicAdd(po + h * 64 + c, alpha * pl[h * 64 + c]);
    }
}

// ---------------- elu + bias -> bf16 activations for the WMMA GEMM ----------------
__global__ void k_elu1(const float* __restrict__ out1, const float* __restrict__ b1,
                       __bf16* __restrict__ h1bf) {
    int node = blockIdx.x, col = threadIdx.x;
    float v = out1[(size_t)node * HC_ + col] + b1[col];
    h1bf[(size_t)node * HC_ + col] = (__bf16)eluf(v);
}

// pre-transpose [Wl2 | Wr2] (256x128) -> Wt[128][256] in bf16 (column-major B)
__global__ void k_prepW2(const float* __restrict__ Wl2, const float* __restrict__ Wr2,
                         __bf16* __restrict__ wt) {
    int n = blockIdx.x;     // 128 output columns
    int k = threadIdx.x;    // 256
    float w = (n < 64) ? Wl2[k * 64 + n] : Wr2[k * 64 + (n - 64)];
    wt[n * 256 + k] = (__bf16)w;
}

// ---------------- layer 2 fused GEMM: [N,256](bf16) x [256,128](bf16) -> xl2|xr2 f32 ----
// one wave = one 16x16 tile; 8 waves/block cover all 128 columns of a 16-row node tile
__global__ void k_gemm2_wmma(const __bf16* __restrict__ h1, const __bf16* __restrict__ wt,
                             float* __restrict__ xl2, float* __restrict__ xr2) {
    int lane = threadIdx.x & 31;
    int wave = threadIdx.x >> 5;            // 0..7 = output column tile
    int m0   = blockIdx.x * 16;             // node tile
    int hi   = lane >> 4;                   // half-wave select
    int l15  = lane & 15;

    v8f acc = {0.f, 0.f, 0.f, 0.f, 0.f, 0.f, 0.f, 0.f};

    // A: lane holds row m0+l15; lo half-wave K{0..7,16..23}, hi half-wave K{8..15,24..31}
    const __bf16* arow = h1 + (size_t)(m0 + l15) * 256 + hi * 8;
    // B: lane holds column wave*16+l15; lo half-wave K{0..7,16..23}, hi half-wave K{8..15,24..31}
    const __bf16* bcol = wt + (size_t)(wave * 16 + l15) * 256 + hi * 8;

    for (int kk = 0; kk < 8; ++kk) {
        int k0 = kk * 32;
        FragBF a, b;
        a.u[0] = *(const uint4*)(arow + k0);
        a.u[1] = *(const uint4*)(arow + k0 + 16);
        b.u[0] = *(const uint4*)(bcol + k0);
        b.u[1] = *(const uint4*)(bcol + k0 + 16);
        acc = __builtin_amdgcn_wmma_f32_16x16x32_bf16(
            false, a.v, false, b.v, (short)0, acc, false, false);
    }

    // C/D layout: VGPR r -> (M = r + 8*hi, N = l15)
    int ncol = wave * 16 + l15;
    for (int r = 0; r < 8; ++r) {
        int row = m0 + r + hi * 8;
        float v = acc[r];
        if (ncol < 64) xl2[(size_t)row * C_ + ncol] = v;
        else           xr2[(size_t)row * C_ + (ncol - 64)] = v;
    }
}

// ---------------- layer 2 edge softmax (1 head, 64 ch) ----------------
__global__ void k_edge2_a(const int* __restrict__ ei,
                          const float* __restrict__ xl, const float* __restrict__ xr,
                          const float* __restrict__ att,
                          float* __restrict__ e2, unsigned* __restrict__ emax) {
    __shared__ float satt[C_];
    if (threadIdx.x < C_) satt[threadIdx.x] = att[threadIdx.x];
    __syncthreads();
    int idx = blockIdx.x * blockDim.x + threadIdx.x;
    if (idx >= ETOT_) return;
    int s, d;
    if (idx < E_) { s = ei[idx]; d = ei[E_ + idx]; } else { s = d = idx - E_; }
    const float* pl = xl + (size_t)s * C_;
    const float* pr = xr + (size_t)d * C_;
    float acc = 0.f;
    for (int c = 0; c < C_; ++c)
        acc += satt[c] * lrelu(pl[c] + pr[c]);
    e2[idx] = acc;
    atomicMax(emax + d, fenc(acc));
}

__global__ void k_edge2_b(const int* __restrict__ ei,
                          float* __restrict__ e2, const unsigned* __restrict__ emax,
                          float* __restrict__ denom) {
    int idx = blockIdx.x * blockDim.x + threadIdx.x;
    if (idx >= ETOT_) return;
    int d = (idx < E_) ? ei[E_ + idx] : (idx - E_);
    float ex = expf(e2[idx] - fdec(emax[d]));
    e2[idx] = ex;
    atomicAdd(denom + d, ex);
}

__global__ void k_edge2_c(const int* __restrict__ ei,
                          const float* __restrict__ e2, const float* __restrict__ denom,
                          const float* __restrict__ xl, float* __restrict__ out) {
    int idx = blockIdx.x * blockDim.x + threadIdx.x;
    if (idx >= ETOT_) return;
    int s, d;
    if (idx < E_) { s = ei[idx]; d = ei[E_ + idx]; } else { s = d = idx - E_; }
    float alpha = e2[idx] / denom[d];
    const float* pl = xl + (size_t)s * C_;
    float* po = out + (size_t)d * C_;
    for (int c = 0; c < C_; ++c)
        atomicAdd(po + c, alpha * pl[c]);
}

__global__ void k_elu2(const float* __restrict__ out2, const float* __restrict__ b2,
                       float* __restrict__ h2) {
    int node = blockIdx.x, col = threadIdx.x;   // 64 threads
    float v = out2[(size_t)node * C_ + col] + b2[col];
    h2[(size_t)node * C_ + col] = eluf(v);
}

// ---------------- mean pool + concat scalars ----------------
__global__ void k_pool(const float* __restrict__ h2,
                       const float* __restrict__ pegs, const float* __restrict__ moves,
                       float* __restrict__ comb) {
    int g = blockIdx.x, c = threadIdx.x;        // 64 threads
    float s = 0.f;
    const float* p = h2 + (size_t)g * NPG_ * C_ + c;
    for (int i = 0; i < NPG_; ++i) s += p[(size_t)i * C_];
    comb[(size_t)g * 66 + c] = s * (1.f / (float)NPG_);
    if (c == 0) {
        comb[(size_t)g * 66 + 64] = pegs[g];
        comb[(size_t)g * 66 + 65] = moves[g];
    }
}

// ---------------- heads ----------------
__global__ void k_head1(const float* __restrict__ comb,
                        const float* __restrict__ vW1, const float* __restrict__ vb1,
                        const float* __restrict__ pW1, const float* __restrict__ pb1,
                        float* __restrict__ hv, float* __restrict__ hp) {
    int g = blockIdx.x, j = threadIdx.x;        // 64 threads
    float av = vb1[j], ap = pb1[j];
    const float* cg = comb + (size_t)g * 66;
    for (int k = 0; k < 66; ++k) {
        float cv = cg[k];
        av += cv * vW1[k * 64 + j];
        ap += cv * pW1[k * 64 + j];
    }
    hv[(size_t)g * 64 + j] = fmaxf(av, 0.f);
    hp[(size_t)g * 64 + j] = fmaxf(ap, 0.f);
}

__global__ void k_head2(const float* __restrict__ hv, const float* __restrict__ hp,
                        const float* __restrict__ vW2, const float* __restrict__ vb2,
                        const float* __restrict__ pW2, const float* __restrict__ pb2,
                        float* __restrict__ out) {
    int g = blockIdx.x, t = threadIdx.x;        // 256 threads
    if (t < NA_) {
        float a = pb2[t];
        const float* h = hp + (size_t)g * 64;
        for (int j = 0; j < 64; ++j) a += h[j] * pW2[j * NA_ + t];
        out[(size_t)g * NA_ + t] = a;
    } else if (t == NA_) {
        float a = vb2[0];
        const float* h = hv + (size_t)g * 64;
        for (int j = 0; j < 64; ++j) a += h[j] * vW2[j];
        out[(size_t)B_ * NA_ + g] = tanhf(a);
    }
}

// ---------------- host launcher ----------------
static inline size_t align256(size_t x) { return (x + 255) & ~(size_t)255; }

extern "C" void kernel_launch(void* const* d_in, const int* in_sizes, int n_in,
                              void* d_out, int out_size, void* d_ws, size_t ws_size,
                              hipStream_t stream) {
    (void)in_sizes; (void)n_in; (void)out_size; (void)ws_size;

    const float* x     = (const float*)d_in[0];
    const int*   ei    = (const int*)d_in[1];
    const float* pegs  = (const float*)d_in[3];
    const float* moves = (const float*)d_in[4];
    const float* Wl1   = (const float*)d_in[5];
    const float* Wr1   = (const float*)d_in[6];
    const float* att1  = (const float*)d_in[7];
    const float* b1    = (const float*)d_in[8];
    const float* Wl2   = (const float*)d_in[9];
    const float* Wr2   = (const float*)d_in[10];
    const float* att2  = (const float*)d_in[11];
    const float* b2    = (const float*)d_in[12];
    const float* vW1   = (const float*)d_in[13];
    const float* vb1   = (const float*)d_in[14];
    const float* vW2   = (const float*)d_in[15];
    const float* vb2   = (const float*)d_in[16];
    const float* pW1   = (const float*)d_in[17];
    const float* pb1   = (const float*)d_in[18];
    const float* pW2   = (const float*)d_in[19];
    const float* pb2   = (const float*)d_in[20];
    float* out = (float*)d_out;

    // bump-allocate workspace (stream-ordered aliasing noted inline)
    char* ws = (char*)d_ws;
    size_t off = 0;
    auto alloc = [&](size_t bytes) -> char* {
        char* p = ws + off; off = align256(off + bytes); return p;
    };
    float*    xl1   = (float*)alloc((size_t)N_ * HC_ * 4);      // dead after edge1_c
    float*    xr1   = (float*)alloc((size_t)N_ * HC_ * 4);      // dead after edge1_a
    float*    e1    = (float*)alloc((size_t)ETOT_ * 4 * 4);     // scores -> exp in place
    unsigned* emax1 = (unsigned*)alloc((size_t)N_ * 4 * 4);
    float*    den1  = (float*)alloc((size_t)N_ * 4 * 4);
    __bf16*   wt    = (__bf16*)alloc((size_t)128 * 256 * 2);
    float*    xl2   = (float*)alloc((size_t)N_ * C_ * 4);       // dead after edge2_c
    float*    xr2   = (float*)alloc((size_t)N_ * C_ * 4);       // dead after edge2_a
    float*    e2    = (float*)alloc((size_t)ETOT_ * 4);
    unsigned* emax2 = (unsigned*)alloc((size_t)N_ * 4);
    float*    den2  = (float*)alloc((size_t)N_ * 4);
    float*    comb  = (float*)alloc((size_t)B_ * 66 * 4);
    float*    hv    = (float*)alloc((size_t)B_ * 64 * 4);
    float*    hp    = (float*)alloc((size_t)B_ * 64 * 4);

    float*  out1 = xr1;              // reuse xr1 storage after pass A (zeroed first)
    __bf16* h1bf = (__bf16*)xl1;     // reuse xl1 storage after pass C
    float*  out2 = xr2;              // reuse xr2 storage after pass A (zeroed first)
    float*  h2   = xl2;              // reuse xl2 storage after pass C

    const int TB = 256;
    const int egrid = (ETOT_ + TB - 1) / TB;

    // ---- layer 1 ----
    k_lin1<<<N_, TB, 0, stream>>>(x, Wl1, Wr1, xl1, xr1);
    k_init_u32<<<(N_ * 4 + TB - 1) / TB, TB, 0, stream>>>(emax1, NEG_INF_ENC, N_ * 4);
    k_edge1_a<<<egrid, TB, 0, stream>>>(ei, xl1, xr1, att1, e1, emax1);
    k_zero_f32<<<((int)((size_t)N_ * HC_) + TB - 1) / TB, TB, 0, stream>>>(out1, N_ * HC_);
    k_zero_f32<<<(N_ * 4 + TB - 1) / TB, TB, 0, stream>>>(den1, N_ * 4);
    k_edge1_b<<<egrid, TB, 0, stream>>>(ei, e1, emax1, den1);
    k_edge1_c<<<egrid, TB, 0, stream>>>(ei, e1, den1, xl1, out1);
    k_elu1<<<N_, TB, 0, stream>>>(out1, b1, h1bf);

    // ---- layer 2 projections via WMMA (bf16 in, f32 acc) ----
    k_prepW2<<<128, 256, 0, stream>>>(Wl2, Wr2, wt);
    k_gemm2_wmma<<<N_ / 16, 256, 0, stream>>>(h1bf, wt, xl2, xr2);

    // ---- layer 2 edge softmax ----
    k_init_u32<<<(N_ + TB - 1) / TB, TB, 0, stream>>>(emax2, NEG_INF_ENC, N_);
    k_edge2_a<<<egrid, TB, 0, stream>>>(ei, xl2, xr2, att2, e2, emax2);
    k_zero_f32<<<(N_ * C_ + TB - 1) / TB, TB, 0, stream>>>(out2, N_ * C_);
    k_zero_f32<<<(N_ + TB - 1) / TB, TB, 0, stream>>>(den2, N_);
    k_edge2_b<<<egrid, TB, 0, stream>>>(ei, e2, emax2, den2);
    k_edge2_c<<<egrid, TB, 0, stream>>>(ei, e2, den2, xl2, out2);
    k_elu2<<<N_, C_, 0, stream>>>(out2, b2, h2);

    // ---- pool + heads ----
    k_pool<<<B_, C_, 0, stream>>>(h2, pegs, moves, comb);
    k_head1<<<B_, 64, 0, stream>>>(comb, vW1, vb1, pW1, pb1, hv, hp);
    k_head2<<<B_, 256, 0, stream>>>(hv, hp, vW2, vb2, pW2, pb2, out);
}
